// MPNN_77360950936126
// MI455X (gfx1250) — matrix operand order repaired and checked by reading, not compiled, not run on hardware
//
#include <hip/hip_runtime.h>
#include <math.h>

#define N_NODES 20000
#define N_EDGES 50000
#define NB      1024
#define IN_DIM  32
#define H       64

typedef _Float16 half16 __attribute__((ext_vector_type(16)));
typedef float    float8 __attribute__((ext_vector_type(8)));
typedef int      i32x4  __attribute__((ext_vector_type(4)));

#define WMMA_F32_F16(a, b, c) \
  __builtin_amdgcn_wmma_f32_16x16x32_f16(false, (a), false, (b), (short)0, (c), false, false)

#if defined(__has_builtin)
#if __has_builtin(__builtin_amdgcn_global_load_async_to_lds_b128)
#define HAVE_ASYNC_B128 1
#endif
#if __has_builtin(__builtin_amdgcn_s_wait_asynccnt)
#define HAVE_WAIT_ASYNC 1
#endif
#endif

#define GAS __attribute__((address_space(1)))
#define LAS __attribute__((address_space(3)))

// 16B global -> LDS copy; async (ASYNCcnt-tracked, no VGPR round trip) when available.
__device__ __forceinline__ void copy_b128_to_lds(const _Float16* g, _Float16* l) {
#if HAVE_ASYNC_B128
  GAS void* gv = (GAS void*)g;   // addrspacecast generic -> global
  LAS void* lv = (LAS void*)l;   // addrspacecast generic -> LDS
  __builtin_amdgcn_global_load_async_to_lds_b128((GAS i32x4*)gv, (LAS i32x4*)lv, 0, 0);
#else
  *(uint4*)l = *(const uint4*)g;
#endif
}
__device__ __forceinline__ void copy_b128_wait() {
#if HAVE_ASYNC_B128
#if HAVE_WAIT_ASYNC
  __builtin_amdgcn_s_wait_asynccnt(0);
#else
  asm volatile("s_wait_asynccnt 0x0" ::: "memory");
#endif
#endif
}

__device__ __forceinline__ float sigmoidf_(float x) { return 1.f / (1.f + expf(-x)); }

__device__ __forceinline__ void atomicMaxF(float* a, float v) {
  if (v >= 0.f) atomicMax((int*)a, __float_as_int(v));
  else          atomicMin((unsigned int*)a, __float_as_uint(v));
}

// A fragment: 16x32 f16, rows M=0..15 in lane%16; lane<16 holds K={k0..k0+7, k0+16..k0+23},
// lane>=16 holds K={k0+8..k0+15, k0+24..k0+31}  (ISA 7.12.2, 16-bit A 16x32)
__device__ __forceinline__ half16 frag_a_lds(const _Float16* src, int ld, int k0) {
  const int l = threadIdx.x & 31;
  const int m = l & 15;
  const int kb = k0 + ((l >> 4) << 3);
  half16 a;
#pragma unroll
  for (int e = 0; e < 8; ++e) a[e] = src[m * ld + kb + e];
#pragma unroll
  for (int e = 0; e < 8; ++e) a[e + 8] = src[m * ld + kb + 16 + e];
  return a;
}

// B fragment: 32x16 f16, N = lane%16; lanes 0-15 hold K={k0..k0+7, k0+16..k0+23}, lanes 16-31 K={k0+8..k0+15, k0+24..k0+31}
__device__ __forceinline__ half16 frag_b_any(const _Float16* src, int ld, int k0, int n0) {
  const int l = threadIdx.x & 31;
  const int n = n0 + (l & 15);
  const int kb = k0 + ((l >> 4) << 3);
  half16 b;
#pragma unroll
  for (int e = 0; e < 16; ++e) b[e] = src[(kb + (e & 7) + ((e >> 3) << 4)) * ld + n];
  return b;
}

// ---------------- weight preprocessing ----------------
__global__ void k_cast_f16(const float* __restrict__ s, _Float16* __restrict__ d, int n) {
  int i = blockIdx.x * blockDim.x + threadIdx.x;
  if (i < n) d[i] = (_Float16)s[i];
}

// d[c*rows + r] = s[r*cols + c]  (transpose + cast; for gru W^T as B operand)
__global__ void k_castT_f16(const float* __restrict__ s, _Float16* __restrict__ d, int rows, int cols) {
  int i = blockIdx.x * blockDim.x + threadIdx.x;
  if (i < rows * cols) {
    int r = i / cols, c = i % cols;
    d[c * rows + r] = (_Float16)s[i];
  }
}

// W2' [4160][64]: row (k*64+h) = We2[k, h*64 + :]; rows 4096..4159 = be2 reshaped (bias block)
__global__ void k_build_w2p(const float* __restrict__ We2, const float* __restrict__ be2,
                            _Float16* __restrict__ w2p) {
  int i = blockIdx.x * blockDim.x + threadIdx.x;
  if (i >= 4160 * H) return;
  int col = i >> 6, o = i & 63;
  int k = col >> 6, h = col & 63;
  float v = (k < H) ? We2[k * (H * H) + h * H + o] : be2[h * H + o];
  w2p[i] = (_Float16)v;
}

// ---------------- lin0: out = relu(x @ W0 + b0), WMMA, one 16-row tile per block ----------------
__global__ void __launch_bounds__(128) k_lin0(const float* __restrict__ x,
                                              const _Float16* __restrict__ W0h,
                                              const float* __restrict__ b0,
                                              float* __restrict__ hout) {
  __shared__ _Float16 xA[16 * IN_DIM];
  const int r0 = blockIdx.x * 16;
  for (int idx = threadIdx.x; idx < 16 * IN_DIM; idx += blockDim.x) {
    int r = idx >> 5, c = idx & 31;
    xA[idx] = (_Float16)x[(r0 + r) * IN_DIM + c];
  }
  __syncthreads();
  const int w = threadIdx.x >> 5;  // N-tile 0..3
  const int l = threadIdx.x & 31;
  float8 acc = {};
  half16 a = frag_a_lds(xA, IN_DIM, 0);
  half16 b = frag_b_any(W0h, H, 0, w * 16);
  acc = WMMA_F32_F16(a, b, acc);
  const int col = w * 16 + (l & 15);
  const int rb = (l >> 4) << 3;
#pragma unroll
  for (int r = 0; r < 8; ++r) {
    float v = acc[r] + b0[col];
    hout[(r0 + rb + r) * H + col] = v > 0.f ? v : 0.f;
  }
}

// ---------------- edge hidden: relu(edge_attr @ We1 + be1) -> f16 [E,64] ----------------
__global__ void k_edge_hidden(const float* __restrict__ ea, const float* __restrict__ We1,
                              const float* __restrict__ be1, _Float16* __restrict__ hid) {
  int i = blockIdx.x * blockDim.x + threadIdx.x;
  if (i >= N_EDGES * H) return;
  int e = i >> 6, o = i & 63;
  float v = be1[o];
#pragma unroll
  for (int j = 0; j < 6; ++j) v += ea[e * 6 + j] * We1[j * H + o];
  hid[i] = (_Float16)(v > 0.f ? v : 0.f);
}

__global__ void k_fill(float* p, int n, float v) {
  int i = blockIdx.x * blockDim.x + threadIdx.x;
  if (i < n) p[i] = v;
}

// ---------------- fused NNConv messages ----------------
// msg tile = P @ W2', P[e, k*64+h] = hid[e,k] * out[src[e],h]   (k==64 -> bias block, hid=1)
// 64 edges per block; each wave owns one 16-col N-tile and 4 M-tiles (4 accumulators).
// W2' K-chunks (32x64 f16 = 4KB) double-buffered in LDS via async global->LDS loads.
__global__ void __launch_bounds__(128) k_messages(const float* __restrict__ hout,
                                                  const _Float16* __restrict__ hid,
                                                  const _Float16* __restrict__ w2p,
                                                  const int* __restrict__ eidx,
                                                  float* __restrict__ agg) {
  __shared__ _Float16 aT[64 * H];       // gathered out[src] (f16)
  __shared__ _Float16 hT[64 * H];       // hidden[e] (f16)
  __shared__ _Float16 bT[2][32 * H];    // double-buffered W2' chunk
  const int e0 = blockIdx.x * 64;
  for (int idx = threadIdx.x; idx < 64 * H; idx += blockDim.x) {
    int e = idx >> 6, c = idx & 63;
    int ge = e0 + e;
    if (ge < N_EDGES) {
      aT[idx] = (_Float16)hout[eidx[ge] * H + c];  // src gather
      hT[idx] = hid[ge * H + c];
    } else {
      aT[idx] = (_Float16)0.f;
      hT[idx] = (_Float16)0.f;
    }
  }
  // prefetch chunk 0
  for (int idx = threadIdx.x; idx < 256; idx += blockDim.x)
    copy_b128_to_lds(w2p + idx * 8, &bT[0][idx * 8]);
  copy_b128_wait();
  __syncthreads();

  const int w = threadIdx.x >> 5;  // N-tile 0..3
  const int l = threadIdx.x & 31;
  const int m16 = l & 15;
  const int cb = (l >> 4) << 3;
  float8 acc0 = {}, acc1 = {}, acc2 = {}, acc3 = {};
  for (int step = 0; step < 130; ++step) {
    const int cur = step & 1;
    if (step + 1 < 130) {  // async prefetch next chunk while computing
      const _Float16* src = w2p + (step + 1) * 32 * H;
      for (int idx = threadIdx.x; idx < 256; idx += blockDim.x)
        copy_b128_to_lds(src + idx * 8, &bT[cur ^ 1][idx * 8]);
    }
    const int kk = step * 32;
    const int k = kk >> 6;   // hidden unit (constant within 32-chunk)
    const int hb = kk & 63;  // 0 or 32
    half16 b = frag_b_any(bT[cur], H, 0, w * 16);
#pragma unroll
    for (int mt = 0; mt < 4; ++mt) {
      const int m = mt * 16 + m16;
      const _Float16 hv = (k < H) ? hT[m * H + k] : (_Float16)1.0f;
      half16 a;
#pragma unroll
      for (int e = 0; e < 8; ++e) a[e] = hv * aT[m * H + hb + cb + e];
#pragma unroll
      for (int e = 0; e < 8; ++e) a[e + 8] = hv * aT[m * H + hb + cb + 16 + e];
      float8& acc = mt == 0 ? acc0 : mt == 1 ? acc1 : mt == 2 ? acc2 : acc3;
      acc = WMMA_F32_F16(a, b, acc);
    }
    copy_b128_wait();
    __syncthreads();
  }
  const int col = w * 16 + m16;
  const int rb = (l >> 4) << 3;
#pragma unroll
  for (int mt = 0; mt < 4; ++mt) {
    float8& acc = mt == 0 ? acc0 : mt == 1 ? acc1 : mt == 2 ? acc2 : acc3;
#pragma unroll
    for (int r = 0; r < 8; ++r) {
      int ge = e0 + mt * 16 + rb + r;
      if (ge < N_EDGES) atomicAdd(&agg[eidx[N_EDGES + ge] * H + col], acc[r]);
    }
  }
}

// ---------------- fused node update: m = relu(agg + h@Wroot + bconv); GRU step ----------------
__global__ void __launch_bounds__(32) k_node_update(float* __restrict__ hout,
                                                    const float* __restrict__ agg,
                                                    const _Float16* __restrict__ WrootH,
                                                    const float* __restrict__ bconv,
                                                    const _Float16* __restrict__ WihT,
                                                    const _Float16* __restrict__ WhhT,
                                                    const float* __restrict__ bih,
                                                    const float* __restrict__ bhh) {
  __shared__ _Float16 hA[16 * H];
  __shared__ _Float16 mA[16 * H];
  __shared__ float gi[16 * 192];
  __shared__ float gh[16 * 192];
  const int r0 = blockIdx.x * 16;
  const int l = threadIdx.x;
  for (int idx = l; idx < 16 * H; idx += 32)
    hA[idx] = (_Float16)hout[(r0 + (idx >> 6)) * H + (idx & 63)];
  __syncthreads();
  const int coln = l & 15;
  const int rb = (l >> 4) << 3;
  // m = relu(agg + h @ Wroot + bconv)
  for (int nt = 0; nt < 4; ++nt) {
    float8 acc = {};
#pragma unroll
    for (int ks = 0; ks < 2; ++ks) {
      half16 a = frag_a_lds(hA, H, ks * 32);
      half16 b = frag_b_any(WrootH, H, ks * 32, nt * 16);
      acc = WMMA_F32_F16(a, b, acc);
    }
    int col = nt * 16 + coln;
#pragma unroll
    for (int r = 0; r < 8; ++r) {
      float v = acc[r] + agg[(r0 + rb + r) * H + col] + bconv[col];
      mA[(rb + r) * H + col] = (_Float16)(v > 0.f ? v : 0.f);
    }
  }
  __syncthreads();
  // gi = m@Wih^T + bih ; gh = h@Whh^T + bhh  (N=192 -> 12 tiles)
  for (int nt = 0; nt < 12; ++nt) {
    float8 ai = {}, ah = {};
#pragma unroll
    for (int ks = 0; ks < 2; ++ks) {
      half16 am = frag_a_lds(mA, H, ks * 32);
      half16 bi = frag_b_any(WihT, 192, ks * 32, nt * 16);
      ai = WMMA_F32_F16(am, bi, ai);
      half16 ahf = frag_a_lds(hA, H, ks * 32);
      half16 bh = frag_b_any(WhhT, 192, ks * 32, nt * 16);
      ah = WMMA_F32_F16(ahf, bh, ah);
    }
    int col = nt * 16 + coln;
#pragma unroll
    for (int r = 0; r < 8; ++r) {
      gi[(rb + r) * 192 + col] = ai[r] + bih[col];
      gh[(rb + r) * 192 + col] = ah[r] + bhh[col];
    }
  }
  __syncthreads();
  for (int idx = l; idx < 16 * H; idx += 32) {
    int row = idx >> 6, c = idx & 63;
    float ir = gi[row * 192 + c], iz = gi[row * 192 + 64 + c], inn = gi[row * 192 + 128 + c];
    float hr = gh[row * 192 + c], hz = gh[row * 192 + 64 + c], hn = gh[row * 192 + 128 + c];
    float rg = sigmoidf_(ir + hr);
    float zg = sigmoidf_(iz + hz);
    float ng = tanhf(inn + rg * hn);
    float hold = hout[(r0 + row) * H + c];
    hout[(r0 + row) * H + c] = (1.f - zg) * ng + zg * hold;
  }
}

// ---------------- Set2Set ----------------
__global__ void k_lstm(const float* __restrict__ qstar, const float* __restrict__ qh_in,
                       float* __restrict__ qh_out, float* __restrict__ qc,
                       const float* __restrict__ wih, const float* __restrict__ whh,
                       const float* __restrict__ bihL, const float* __restrict__ bhhL) {
  int i = blockIdx.x * blockDim.x + threadIdx.x;
  if (i >= NB * H) return;
  int b = i >> 6, j = i & 63;
  float g[4];
#pragma unroll
  for (int gx = 0; gx < 4; ++gx) {
    int row = gx * H + j;
    float v = bihL[row] + bhhL[row];
    const float* wr = wih + row * 2 * H;
    const float* qs = qstar + b * 2 * H;
    for (int k2 = 0; k2 < 2 * H; ++k2) v += qs[k2] * wr[k2];
    const float* hr = whh + row * H;
    const float* qv = qh_in + b * H;
    for (int k2 = 0; k2 < H; ++k2) v += qv[k2] * hr[k2];
    g[gx] = v;
  }
  float ig = sigmoidf_(g[0]), fg = sigmoidf_(g[1]), gg = tanhf(g[2]), og = sigmoidf_(g[3]);
  float c = fg * qc[i] + ig * gg;
  qc[i] = c;
  qh_out[i] = og * tanhf(c);
}

__global__ void k_score(const float* __restrict__ hout, const float* __restrict__ qh,
                        const int* __restrict__ batch, float* __restrict__ esc,
                        float* __restrict__ emax) {
  int n = blockIdx.x * blockDim.x + threadIdx.x;
  if (n >= N_NODES) return;
  int b = batch[n];
  float v = 0.f;
  for (int c = 0; c < H; ++c) v += hout[n * H + c] * qh[b * H + c];
  esc[n] = v;
  atomicMaxF(&emax[b], v);
}

__global__ void k_expsum(const float* __restrict__ esc, const int* __restrict__ batch,
                         const float* __restrict__ emax, float* __restrict__ aw,
                         float* __restrict__ asum) {
  int n = blockIdx.x * blockDim.x + threadIdx.x;
  if (n >= N_NODES) return;
  int b = batch[n];
  float a = expf(esc[n] - emax[b]);
  aw[n] = a;
  atomicAdd(&asum[b], a);
}

__global__ void k_weighted(const float* __restrict__ hout, const int* __restrict__ batch,
                           const float* __restrict__ aw, const float* __restrict__ asum,
                           float* __restrict__ rbuf) {
  int n = blockIdx.x * blockDim.x + threadIdx.x;
  if (n >= N_NODES) return;
  int b = batch[n];
  float w = aw[n] / asum[b];
  for (int c = 0; c < H; ++c) atomicAdd(&rbuf[b * H + c], w * hout[n * H + c]);
}

__global__ void k_qstar(const float* __restrict__ qh, const float* __restrict__ rbuf,
                        float* __restrict__ qstar) {
  int i = blockIdx.x * blockDim.x + threadIdx.x;
  if (i >= NB * 2 * H) return;
  int b = i >> 7, c = i & 127;
  qstar[i] = (c < H) ? qh[b * H + c] : rbuf[b * H + (c - H)];
}

// ---------------- readout MLP: one block per graph ----------------
__global__ void __launch_bounds__(64) k_readout(const float* __restrict__ qstar,
                                                const float* __restrict__ t,
                                                const float* __restrict__ p,
                                                const float* __restrict__ W1,
                                                const float* __restrict__ b1,
                                                const float* __restrict__ W2,
                                                const float* __restrict__ b2,
                                                const float* __restrict__ W3,
                                                const float* __restrict__ b3,
                                                float* __restrict__ out) {
  __shared__ float feat[130];
  __shared__ float y1[64];
  __shared__ float y2[64];
  int b = blockIdx.x, j = threadIdx.x;
  for (int idx = j; idx < 130; idx += 64)
    feat[idx] = idx < 128 ? qstar[b * 128 + idx] : (idx == 128 ? t[b] : p[b]);
  __syncthreads();
  float v = b1[j];
  for (int i2 = 0; i2 < 130; ++i2) v += feat[i2] * W1[i2 * 64 + j];
  y1[j] = v > 0.f ? v : 0.f;
  __syncthreads();
  v = b2[j];
  for (int i2 = 0; i2 < 64; ++i2) v += y1[i2] * W2[i2 * 64 + j];
  y2[j] = v > 0.f ? v : 0.f;
  __syncthreads();
  if (j == 0) {
    float y = b3[0];
    for (int i2 = 0; i2 < 64; ++i2) y += y2[i2] * W3[i2];
    out[b] = y;
  }
}

extern "C" void kernel_launch(void* const* d_in, const int* in_sizes, int n_in,
                              void* d_out, int out_size, void* d_ws, size_t ws_size,
                              hipStream_t stream) {
  const float* x    = (const float*)d_in[0];
  const int*   eidx = (const int*)d_in[1];
  const float* ea   = (const float*)d_in[2];
  const int*   batch= (const int*)d_in[3];
  const float* t    = (const float*)d_in[4];
  const float* p    = (const float*)d_in[5];
  const float* W0   = (const float*)d_in[7];
  const float* b0   = (const float*)d_in[8];
  const float* We1  = (const float*)d_in[9];
  const float* be1  = (const float*)d_in[10];
  const float* We2  = (const float*)d_in[11];
  const float* be2  = (const float*)d_in[12];
  const float* Wroot= (const float*)d_in[13];
  const float* bconv= (const float*)d_in[14];
  const float* gwih = (const float*)d_in[15];
  const float* gwhh = (const float*)d_in[16];
  const float* gbih = (const float*)d_in[17];
  const float* gbhh = (const float*)d_in[18];
  const float* lwih = (const float*)d_in[19];
  const float* lwhh = (const float*)d_in[20];
  const float* lbih = (const float*)d_in[21];
  const float* lbhh = (const float*)d_in[22];
  const float* W1   = (const float*)d_in[23];
  const float* b1   = (const float*)d_in[24];
  const float* W2   = (const float*)d_in[25];
  const float* b2   = (const float*)d_in[26];
  const float* W3   = (const float*)d_in[27];
  const float* b3   = (const float*)d_in[28];
  float* out = (float*)d_out;
  (void)in_sizes; (void)n_in; (void)out_size; (void)ws_size;

  char* base = (char*)d_ws;
  size_t off = 0;
  auto carve = [&](size_t bytes) -> char* {
    off = (off + 255) & ~(size_t)255;
    char* pp = base + off;
    off += bytes;
    return pp;
  };
  float*    hout  = (float*)carve((size_t)N_NODES * H * 4);
  float*    agg   = (float*)carve((size_t)N_NODES * H * 4);
  _Float16* hid   = (_Float16*)carve((size_t)N_EDGES * H * 2);
  _Float16* w2p   = (_Float16*)carve((size_t)4160 * H * 2);
  _Float16* W0h   = (_Float16*)carve((size_t)IN_DIM * H * 2);
  _Float16* WrootH= (_Float16*)carve((size_t)H * H * 2);
  _Float16* WihT  = (_Float16*)carve((size_t)H * 192 * 2);
  _Float16* WhhT  = (_Float16*)carve((size_t)H * 192 * 2);
  float*    qhA   = (float*)carve((size_t)NB * H * 4);
  float*    qhB   = (float*)carve((size_t)NB * H * 4);
  float*    qc    = (float*)carve((size_t)NB * H * 4);
  float*    qstar = (float*)carve((size_t)NB * 2 * H * 4);
  float*    esc   = (float*)carve((size_t)N_NODES * 4);
  float*    aw    = (float*)carve((size_t)N_NODES * 4);
  float*    emax  = (float*)carve((size_t)NB * 4);
  float*    asum  = (float*)carve((size_t)NB * 4);
  float*    rbuf  = (float*)carve((size_t)NB * H * 4);

  const int T = 256;
  // weight prep
  k_cast_f16<<<(IN_DIM * H + T - 1) / T, T, 0, stream>>>(W0, W0h, IN_DIM * H);
  k_cast_f16<<<(H * H + T - 1) / T, T, 0, stream>>>(Wroot, WrootH, H * H);
  k_castT_f16<<<(192 * H + T - 1) / T, T, 0, stream>>>(gwih, WihT, 192, H);
  k_castT_f16<<<(192 * H + T - 1) / T, T, 0, stream>>>(gwhh, WhhT, 192, H);
  k_build_w2p<<<(4160 * H + T - 1) / T, T, 0, stream>>>(We2, be2, w2p);
  // encoders
  k_lin0<<<N_NODES / 16, 128, 0, stream>>>(x, W0h, b0, hout);
  k_edge_hidden<<<(N_EDGES * H + T - 1) / T, T, 0, stream>>>(ea, We1, be1, hid);
  // 4 message-passing rounds
  for (int it = 0; it < 4; ++it) {
    k_fill<<<(N_NODES * H + T - 1) / T, T, 0, stream>>>(agg, N_NODES * H, 0.f);
    k_messages<<<(N_EDGES + 63) / 64, 128, 0, stream>>>(hout, hid, w2p, eidx, agg);
    k_node_update<<<N_NODES / 16, 32, 0, stream>>>(hout, agg, WrootH, bconv, WihT, WhhT, gbih, gbhh);
  }
  // Set2Set (3 steps)
  k_fill<<<(NB * H + T - 1) / T, T, 0, stream>>>(qhA, NB * H, 0.f);
  k_fill<<<(NB * H + T - 1) / T, T, 0, stream>>>(qc, NB * H, 0.f);
  k_fill<<<(NB * 2 * H + T - 1) / T, T, 0, stream>>>(qstar, NB * 2 * H, 0.f);
  float* qr = qhA;
  float* qw = qhB;
  for (int s = 0; s < 3; ++s) {
    k_lstm<<<(NB * H + T - 1) / T, T, 0, stream>>>(qstar, qr, qw, qc, lwih, lwhh, lbih, lbhh);
    k_fill<<<(NB + T - 1) / T, T, 0, stream>>>(emax, NB, -1e30f);
    k_fill<<<(NB + T - 1) / T, T, 0, stream>>>(asum, NB, 0.f);
    k_fill<<<(NB * H + T - 1) / T, T, 0, stream>>>(rbuf, NB * H, 0.f);
    k_score<<<(N_NODES + T - 1) / T, T, 0, stream>>>(hout, qw, batch, esc, emax);
    k_expsum<<<(N_NODES + T - 1) / T, T, 0, stream>>>(esc, batch, emax, aw, asum);
    k_weighted<<<(N_NODES + T - 1) / T, T, 0, stream>>>(hout, batch, aw, asum, rbuf);
    k_qstar<<<(NB * 2 * H + T - 1) / T, T, 0, stream>>>(qw, rbuf, qstar);
    float* tmp = qr; qr = qw; qw = tmp;
  }
  k_readout<<<NB, 64, 0, stream>>>(qstar, t, p, W1, b1, W2, b2, W3, b3, out);
}
